// SimpleLSTM_29283087024347
// MI455X (gfx1250) — matrix operand order repaired and checked by reading, not compiled, not run on hardware
//
#include <hip/hip_runtime.h>

typedef _Float16 v16h __attribute__((ext_vector_type(16)));
typedef _Float16 v8h  __attribute__((ext_vector_type(8)));
typedef float    v8f  __attribute__((ext_vector_type(8)));

#define LOG2E_ 1.4426950408889634f

__device__ __forceinline__ float fast_sigmoid(float x) {
  return __builtin_amdgcn_rcpf(1.0f + __builtin_amdgcn_exp2f(-x * LOG2E_));
}
__device__ __forceinline__ float fast_tanh(float x) {
  return 2.0f * __builtin_amdgcn_rcpf(1.0f + __builtin_amdgcn_exp2f(-2.0f * LOG2E_ * x)) - 1.0f;
}

// Build a 16-bit A-fragment (16x32, MxK) for one wave from an LDS row-major
// [16][HP_] f16 matrix. Contiguous K order: lanes 0-15 hold K=0..15,
// lanes 16-31 hold K=16..31 (kb = 0 or 8 selects the half).
__device__ __forceinline__ v16h afrag(const _Float16* row, int koff, int kb) {
  v8h a0 = *(const v8h*)(row + koff + 2 * kb);
  v8h a1 = *(const v8h*)(row + koff + 2 * kb + 8);
  return __builtin_shufflevector(a0, a1, 0,1,2,3,4,5,6,7,8,9,10,11,12,13,14,15);
}

#define WMMA_F16(A, B, C) \
  __builtin_amdgcn_wmma_f32_16x16x32_f16(false, (A), false, (B), (short)0, (C), false, false)

constexpr int T_ = 256, F_ = 32, H_ = 64, G_ = 256; // G_ = 4*H
constexpr int HP_ = 72;                             // padded LDS row stride (halfs), 144B = 16B-aligned
constexpr int WAVES_ = 4;

__global__ __launch_bounds__(WAVES_ * 32, 1)
void lstm2_wmma_kernel(const float* __restrict__ x,
                       const float* __restrict__ wih0, const float* __restrict__ whh0,
                       const float* __restrict__ bih0, const float* __restrict__ bhh0,
                       const float* __restrict__ wih1, const float* __restrict__ whh1,
                       const float* __restrict__ bih1, const float* __restrict__ bhh1,
                       const float* __restrict__ lng,  const float* __restrict__ lnb,
                       const float* __restrict__ w1,   const float* __restrict__ b1,
                       const float* __restrict__ w2,   const float* __restrict__ b2,
                       float* __restrict__ out)
{
  // f16 weights staged once; B-fragment = one contiguous 32B LDS read per lane.
  __shared__ __align__(32) _Float16 s_wih0[G_ * F_];   // 16 KB
  __shared__ __align__(32) _Float16 s_whh0[G_ * H_];   // 32 KB
  __shared__ __align__(32) _Float16 s_wih1[G_ * H_];   // 32 KB
  __shared__ __align__(32) _Float16 s_whh1[G_ * H_];   // 32 KB
  __shared__ float s_b0[G_], s_b1[G_];                 // 2 KB
  __shared__ __align__(32) _Float16 s_h0[WAVES_][16 * HP_]; // per-wave h scratch
  __shared__ __align__(32) _Float16 s_h1[WAVES_][16 * HP_];

  const int tid = threadIdx.x;
  for (int i = tid; i < G_ * F_; i += blockDim.x) s_wih0[i] = (_Float16)wih0[i];
  for (int i = tid; i < G_ * H_; i += blockDim.x) {
    s_whh0[i] = (_Float16)whh0[i];
    s_wih1[i] = (_Float16)wih1[i];
    s_whh1[i] = (_Float16)whh1[i];
  }
  for (int i = tid; i < G_; i += blockDim.x) {
    s_b0[i] = bih0[i] + bhh0[i];
    s_b1[i] = bih1[i] + bhh1[i];
  }
  __syncthreads();

  const int w  = tid >> 5;
  const int l  = tid & 31;
  const int m  = l & 15;            // row (A layout) / column (C layout) owned by this lane
  const int kb = (l < 16) ? 0 : 8;  // A-fragment K sub-offset
  const int bo = (l < 16) ? 0 : 16; // B-fragment K sub-offset
  const int r0 = (l < 16) ? 0 : 8;  // C-layout M base for this lane half
  const int batch0 = blockIdx.x * (WAVES_ * 16) + w * 16;

  _Float16* h0w = s_h0[w];
  _Float16* h1w = s_h1[w];
  const float* xr = x + (size_t)(batch0 + m) * (T_ * F_);

  v16h h0A0, h0A1, h1A0, h1A1;
  #pragma unroll
  for (int e = 0; e < 16; ++e) {
    h0A0[e] = (_Float16)0.f; h0A1[e] = (_Float16)0.f;
    h1A0[e] = (_Float16)0.f; h1A1[e] = (_Float16)0.f;
  }
  v8f c0a[4], c1a[4];
  #pragma unroll
  for (int j = 0; j < 4; ++j) {
    #pragma unroll
    for (int e = 0; e < 8; ++e) { c0a[j][e] = 0.f; c1a[j][e] = 0.f; }
  }

  #pragma unroll 1
  for (int t = 0; t < T_; ++t) {
    const float* xt = xr + t * F_;
    __builtin_prefetch(xt + F_, 0, 0);  // next timestep row -> global_prefetch_b8

    v16h xa;  // A-fragment of x_t [16 x 32], converted f32 -> f16
    {
      float4 q0 = *(const float4*)(xt + 2 * kb);
      float4 q1 = *(const float4*)(xt + 2 * kb + 4);
      float4 q2 = *(const float4*)(xt + 2 * kb + 8);
      float4 q3 = *(const float4*)(xt + 2 * kb + 12);
      xa[0]=(_Float16)q0.x;  xa[1]=(_Float16)q0.y;  xa[2]=(_Float16)q0.z;  xa[3]=(_Float16)q0.w;
      xa[4]=(_Float16)q1.x;  xa[5]=(_Float16)q1.y;  xa[6]=(_Float16)q1.z;  xa[7]=(_Float16)q1.w;
      xa[8]=(_Float16)q2.x;  xa[9]=(_Float16)q2.y;  xa[10]=(_Float16)q2.z; xa[11]=(_Float16)q2.w;
      xa[12]=(_Float16)q3.x; xa[13]=(_Float16)q3.y; xa[14]=(_Float16)q3.z; xa[15]=(_Float16)q3.w;
    }

    // ---------------- layer 0: gates = b + x_t @ Wih0^T + h0 @ Whh0^T ----------------
    #pragma unroll
    for (int j = 0; j < 4; ++j) {          // hidden 16-column tile
      v8f g[4];
      #pragma unroll
      for (int gi = 0; gi < 4; ++gi) {     // i, f, g, o
        const int nrow = gi * 64 + j * 16 + m;
        const float bv = s_b0[nrow];
        v8f acc = { bv, bv, bv, bv, bv, bv, bv, bv };
        acc = WMMA_F16(xa,   *(const v16h*)(s_wih0 + nrow * F_ + bo),        acc);
        acc = WMMA_F16(h0A0, *(const v16h*)(s_whh0 + nrow * H_ + bo),        acc);
        acc = WMMA_F16(h0A1, *(const v16h*)(s_whh0 + nrow * H_ + 32 + bo),   acc);
        g[gi] = acc;
      }
      const int col = j * 16 + m;
      #pragma unroll
      for (int v = 0; v < 8; ++v) {        // per-lane cell update (same C-layout slot)
        float iv = fast_sigmoid(g[0][v]);
        float fv = fast_sigmoid(g[1][v]);
        float gv = fast_tanh(g[2][v]);
        float ov = fast_sigmoid(g[3][v]);
        float cv = fv * c0a[j][v] + iv * gv;
        c0a[j][v] = cv;
        h0w[(r0 + v) * HP_ + col] = (_Float16)(ov * fast_tanh(cv));
      }
    }
    v16h h0n0 = afrag(h0w + m * HP_, 0,  kb);   // new h0 as A-fragments
    v16h h0n1 = afrag(h0w + m * HP_, 32, kb);

    // ---------------- layer 1: gates = b + h0_new @ Wih1^T + h1 @ Whh1^T ----------------
    #pragma unroll
    for (int j = 0; j < 4; ++j) {
      v8f g[4];
      #pragma unroll
      for (int gi = 0; gi < 4; ++gi) {
        const int nrow = gi * 64 + j * 16 + m;
        const float bv = s_b1[nrow];
        v8f acc = { bv, bv, bv, bv, bv, bv, bv, bv };
        acc = WMMA_F16(h0n0, *(const v16h*)(s_wih1 + nrow * H_ + bo),        acc);
        acc = WMMA_F16(h0n1, *(const v16h*)(s_wih1 + nrow * H_ + 32 + bo),   acc);
        acc = WMMA_F16(h1A0, *(const v16h*)(s_whh1 + nrow * H_ + bo),        acc);
        acc = WMMA_F16(h1A1, *(const v16h*)(s_whh1 + nrow * H_ + 32 + bo),   acc);
        g[gi] = acc;
      }
      const int col = j * 16 + m;
      #pragma unroll
      for (int v = 0; v < 8; ++v) {
        float iv = fast_sigmoid(g[0][v]);
        float fv = fast_sigmoid(g[1][v]);
        float gv = fast_tanh(g[2][v]);
        float ov = fast_sigmoid(g[3][v]);
        float cv = fv * c1a[j][v] + iv * gv;
        c1a[j][v] = cv;
        h1w[(r0 + v) * HP_ + col] = (_Float16)(ov * fast_tanh(cv));
      }
    }
    h1A0 = afrag(h1w + m * HP_, 0,  kb);
    h1A1 = afrag(h1w + m * HP_, 32, kb);
    h0A0 = h0n0; h0A1 = h0n1;
  }

  // ---------------- epilogue: LayerNorm + MLP on h1[T-1]; lane m owns row m ----------------
  if (l < 16) {
    _Float16* hr = h1w + m * HP_;
    float s = 0.f, ss = 0.f;
    #pragma unroll 8
    for (int c = 0; c < H_; ++c) { float v = (float)hr[c]; s += v; ss += v * v; }
    const float mu  = s * (1.0f / H_);
    const float var = ss * (1.0f / H_) - mu * mu;
    const float rs  = rsqrtf(var + 1e-5f);
    #pragma unroll 8
    for (int c = 0; c < H_; ++c) {
      float v = ((float)hr[c] - mu) * rs * lng[c] + lnb[c];
      hr[c] = (_Float16)v;   // reuse scratch to hold LN output
    }
    float y = b2[0];
    for (int k = 0; k < H_; ++k) {
      float acc = b1[k];
      const float* wr = w1 + k * H_;
      #pragma unroll 8
      for (int c = 0; c < H_; ++c) acc += (float)hr[c] * wr[c];
      y += fmaxf(acc, 0.f) * w2[k];
    }
    out[batch0 + m] = y;
  }
}

extern "C" void kernel_launch(void* const* d_in, const int* in_sizes, int n_in,
                              void* d_out, int out_size, void* d_ws, size_t ws_size,
                              hipStream_t stream) {
  const float* x    = (const float*)d_in[0];
  const float* wih0 = (const float*)d_in[1];
  const float* whh0 = (const float*)d_in[2];
  const float* bih0 = (const float*)d_in[3];
  const float* bhh0 = (const float*)d_in[4];
  const float* wih1 = (const float*)d_in[5];
  const float* whh1 = (const float*)d_in[6];
  const float* bih1 = (const float*)d_in[7];
  const float* bhh1 = (const float*)d_in[8];
  const float* lng  = (const float*)d_in[9];
  const float* lnb  = (const float*)d_in[10];
  const float* w1   = (const float*)d_in[11];
  const float* b1   = (const float*)d_in[12];
  const float* w2   = (const float*)d_in[13];
  const float* b2   = (const float*)d_in[14];

  const int B = in_sizes[0] / (T_ * F_);     // 4096
  const int blocks = B / (WAVES_ * 16);      // 64 workgroups of 4 waves
  lstm2_wmma_kernel<<<blocks, WAVES_ * 32, 0, stream>>>(
      x, wih0, whh0, bih0, bhh0, wih1, whh1, bih1, bhh1,
      lng, lnb, w1, b1, w2, b2, (float*)d_out);
}
